// CharRNN_90366111907901
// MI455X (gfx1250) — matrix operand order repaired and checked by reading, not compiled, not run on hardware
//
#include <hip/hip_runtime.h>
#include <hip/hip_bf16.h>

typedef __bf16 bf16_t;
typedef __attribute__((ext_vector_type(8)))  __bf16 bf16x8;
typedef __attribute__((ext_vector_type(16))) __bf16 bf16x16;
typedef __attribute__((ext_vector_type(8)))  float  f32x8;

namespace {
constexpr int VOC  = 32;
constexpr int HID  = 1024;
constexpr int NLAY = 3;
constexpr int BAT  = 256;
constexpr int SEQ  = 128;
constexpr int G4H  = 4 * HID;   // 4096
}

// -------- gfx1250 async global->LDS path (guarded; sync fallback otherwise) --------
#if defined(__has_builtin)
#if __has_builtin(__builtin_amdgcn_global_load_async_to_lds_b128) && \
    __has_builtin(__builtin_amdgcn_s_wait_asynccnt)
#define USE_ASYNC_LDS 1
#endif
#endif

#if defined(USE_ASYNC_LDS)
// builtin expects: (int __vector(4) __device__ *, int __vector(4) __shared__ *, imm, imm)
typedef int v4i_gcc __attribute__((vector_size(16)));
typedef __attribute__((address_space(1))) v4i_gcc as1_v4i;
typedef __attribute__((address_space(3))) v4i_gcc as3_v4i;
#endif

// ---------------- WMMA fragment helpers (bf16, 16x16x32) ----------------
// A (16x32, MxK) per-lane layout: lane&15 = M row; lane>>4 selects K-subgroup.
__device__ __forceinline__ bf16x16 load_a_frag(const bf16_t* __restrict__ base,
                                               int lda, int mbase, int k, int lane) {
  const int row  = mbase + (lane & 15);
  const int half = lane >> 4;
  const bf16_t* p = base + (size_t)row * lda + k + half * 16;
  bf16x8 lo = *(const bf16x8*)(p);        // K = half*16 .. +7
  bf16x8 hi = *(const bf16x8*)(p + 8);    // K = half*16 + 8 .. +15
  return __builtin_shufflevector(lo, hi, 0,1,2,3,4,5,6,7,8,9,10,11,12,13,14,15);
}

// B (32x16, KxN): lane&15 = N col; lane<16: K=0..15, lane>=16: K=16..31.
// Fragment data staged in LDS "fragment order": frag f at f*1024 + lane*32 bytes.
__device__ __forceinline__ bf16x16 lds_b_frag(const char* buf, int f, int lane) {
  const bf16_t* p = (const bf16_t*)(buf + f * 1024 + lane * 32);
  bf16x8 lo = *(const bf16x8*)(p);
  bf16x8 hi = *(const bf16x8*)(p + 8);
  return __builtin_shufflevector(lo, hi, 0,1,2,3,4,5,6,7,8,9,10,11,12,13,14,15);
}

// direct-from-global B fragment (used by the small logits head)
__device__ __forceinline__ bf16x16 load_b_frag(const bf16_t* __restrict__ W,
                                               int ldb, int nbase, int k, int lane) {
  const int col  = nbase + (lane & 15);
  const int half = lane >> 4;
  const bf16_t* p = W + (size_t)col * ldb + k + half * 16;
  bf16x8 lo = *(const bf16x8*)(p);
  bf16x8 hi = *(const bf16x8*)(p + 8);
  return __builtin_shufflevector(lo, hi, 0,1,2,3,4,5,6,7,8,9,10,11,12,13,14,15);
}

__device__ __forceinline__ f32x8 wmma_bf16(bf16x16 a, bf16x16 b, f32x8 c) {
  return __builtin_amdgcn_wmma_f32_16x16x32_bf16(false, a, false, b, (short)0, c, false, false);
}

__device__ __forceinline__ float sigmoidf_(float x) { return 1.f / (1.f + __expf(-x)); }

// ---------------- small prep kernels ----------------
__global__ void f32_to_bf16_kernel(const float* __restrict__ src, bf16_t* __restrict__ dst, int n) {
  for (int i = blockIdx.x * blockDim.x + threadIdx.x; i < n; i += gridDim.x * blockDim.x)
    dst[i] = (bf16_t)src[i];
}

__global__ void zero_f32_kernel(float* __restrict__ dst, int n) {
  for (int i = blockIdx.x * blockDim.x + threadIdx.x; i < n; i += gridDim.x * blockDim.x)
    dst[i] = 0.f;
}

// masked one-hot: x0[t][b][v] = (t < len[b] && tokens[b][t]==v) ? 1 : 0   (bf16)
__global__ void onehot_kernel(const int* __restrict__ tokens, const int* __restrict__ lengths,
                              bf16_t* __restrict__ x0) {
  const int total = SEQ * BAT * VOC;   // 2^20
  for (int i = blockIdx.x * blockDim.x + threadIdx.x; i < total; i += gridDim.x * blockDim.x) {
    const int v = i & (VOC - 1);
    const int b = (i >> 5) & (BAT - 1);
    const int t = i >> 13;             // / (BAT*VOC)
    const int tok = tokens[b * SEQ + t];
    const float val = (t < lengths[b] && tok == v) ? 1.f : 0.f;
    x0[i] = (bf16_t)val;
  }
}

// ---------------- fused LSTM step ----------------
// Block = 8 waves sharing one hidden-column group n0 (all waves consume the SAME
// weight fragments, staged once per block into LDS), covering 8 batch tiles.
// Grid: 64 blocks; block b -> n0 = (b>>1)*32, mtile = (b&1)*8 + wave.
// K-tiles: nkt_x tiles of W_ih (input proj) then, if t>0, 32 tiles of W_hh (recurrent).
__global__ __launch_bounds__(256) void lstm_step_kernel(
    const bf16_t* __restrict__ x_in,  // [BAT, lda_x] input at this timestep
    int lda_x, int nkt_x,
    const bf16_t* __restrict__ W_ih,  // [4H, lda_x] row-major bf16
    const bf16_t* __restrict__ W_hh,  // [4H, HID]   row-major bf16
    const float*  __restrict__ bias,  // [4H]
    bf16_t* __restrict__ hs,          // [SEQ, BAT, HID] this layer's output (t-1 read, t write)
    float*  __restrict__ h_state,     // [BAT, HID] fp32
    float*  __restrict__ c_state,     // [BAT, HID] fp32
    const int* __restrict__ lengths, int t)
{
  __shared__ __align__(16) char ldsbuf[2][8 * 1024];   // 8 B-frags/ktile, double buffered

  const int tid   = threadIdx.x;
  const int lane  = tid & 31;
  const int wave  = tid >> 5;
  const int n0    = (blockIdx.x >> 1) * 32;            // hidden col group (shared by block)
  const int mbase = (((blockIdx.x & 1) << 3) + wave) * 16;  // batch rows (per wave)

  const bf16_t* h_prev = hs + (size_t)(t - 1) * BAT * HID;
  const int nkt_h  = (t > 0) ? 32 : 0;
  const int total  = nkt_x + nkt_h;

  // this thread's staging role: fragment f = tid>>5, within-frag lane = tid&31
  const int sf    = tid >> 5;
  const int sgate = sf >> 1, sj = sf & 1;
  const int scol  = sgate * HID + n0 + sj * 16 + (lane & 15);
  const int skoff = (lane >> 4) * 16;
  char* const sdst = &ldsbuf[0][0] + sf * 1024 + lane * 32; // buffer 0 offset; +8192 for buf1

  auto stage_src = [&](int i) -> const bf16_t* {
    if (i < nkt_x) return W_ih + (size_t)scol * lda_x + i * 32 + skoff;
    return W_hh + (size_t)scol * HID + (i - nkt_x) * 32 + skoff;
  };
  auto load_a_tile = [&](int i) -> bf16x16 {
    const bool xpart = (i < nkt_x);
    const bf16_t* abase = xpart ? x_in : h_prev;
    const int alda = xpart ? lda_x : HID;
    const int ak   = (xpart ? i : i - nkt_x) * 32;
    return load_a_frag(abase, alda, mbase, ak, lane);
  };

  f32x8 acc[4][2];
  #pragma unroll
  for (int g = 0; g < 4; ++g)
    #pragma unroll
    for (int j = 0; j < 2; ++j)
      acc[g][j] = f32x8{0.f,0.f,0.f,0.f,0.f,0.f,0.f,0.f};

#if defined(USE_ASYNC_LDS)
  // ---- async global->LDS double-buffered pipeline; A double-buffered in VGPRs ----
  {
    const bf16_t* g0 = stage_src(0);
    __builtin_amdgcn_global_load_async_to_lds_b128((as1_v4i*)g0, (as3_v4i*)sdst, 0, 0);
    __builtin_amdgcn_global_load_async_to_lds_b128((as1_v4i*)g0, (as3_v4i*)sdst, 16, 0);
  }
  bf16x16 a_cur = load_a_tile(0);
  for (int i = 0; i < total; ++i) {
    if (i + 1 < total) {
      const bf16_t* gn = stage_src(i + 1);
      char* dn = sdst + ((i + 1) & 1) * 8192;
      __builtin_amdgcn_global_load_async_to_lds_b128((as1_v4i*)gn, (as3_v4i*)dn, 0, 0);
      __builtin_amdgcn_global_load_async_to_lds_b128((as1_v4i*)gn, (as3_v4i*)dn, 16, 0);
      __builtin_amdgcn_s_wait_asynccnt(2);   // tile i complete, tile i+1 may fly
    } else {
      __builtin_amdgcn_s_wait_asynccnt(0);
    }
    __syncthreads();

    bf16x16 a_next;
    if (i + 1 < total) a_next = load_a_tile(i + 1);  // issue now, consume next iter

    const char* buf = ldsbuf[i & 1];
    bf16x16 B[8];
    #pragma unroll
    for (int f = 0; f < 8; ++f) B[f] = lds_b_frag(buf, f, lane);   // 16 ds_loads, one wait
    #pragma unroll
    for (int f = 0; f < 8; ++f)
      acc[f >> 1][f & 1] = wmma_bf16(a_cur, B[f], acc[f >> 1][f & 1]);  // independent WMMAs
    __syncthreads();
    a_cur = a_next;
  }
#else
  // ---- sync fallback: register-staged global->LDS double buffer ----
  uint4 r0, r1;
  {
    const bf16_t* g0 = stage_src(0);
    r0 = *(const uint4*)(g0);
    r1 = *(const uint4*)(g0 + 8);
  }
  bf16x16 a_cur = load_a_tile(0);
  for (int i = 0; i < total; ++i) {
    char* dcur = sdst + (i & 1) * 8192;
    *(uint4*)(dcur)      = r0;
    *(uint4*)(dcur + 16) = r1;
    __syncthreads();

    bf16x16 a_next;
    if (i + 1 < total) {   // issue next tile's loads; overlap with compute
      const bf16_t* gn = stage_src(i + 1);
      r0 = *(const uint4*)(gn);
      r1 = *(const uint4*)(gn + 8);
      a_next = load_a_tile(i + 1);
    }

    const char* buf = ldsbuf[i & 1];
    bf16x16 B[8];
    #pragma unroll
    for (int f = 0; f < 8; ++f) B[f] = lds_b_frag(buf, f, lane);
    #pragma unroll
    for (int f = 0; f < 8; ++f)
      acc[f >> 1][f & 1] = wmma_bf16(a_cur, B[f], acc[f >> 1][f & 1]);
    __syncthreads();
    a_cur = a_next;
  }
#endif

  // ---- fused pointwise LSTM cell ----
  const int half = lane >> 4;
  const int ncol = lane & 15;
  bf16_t* hs_t = hs + (size_t)t * BAT * HID;
  #pragma unroll
  for (int j = 0; j < 2; ++j) {
    const int hcol = n0 + j * 16 + ncol;
    const float bi = bias[0 * HID + hcol];
    const float bf = bias[1 * HID + hcol];
    const float bg = bias[2 * HID + hcol];
    const float bo = bias[3 * HID + hcol];
    #pragma unroll
    for (int r = 0; r < 8; ++r) {
      const int b = mbase + half * 8 + r;      // batch row for this C/D element
      const size_t idx = (size_t)b * HID + hcol;
      const bool m = (t < lengths[b]);
      const float iv = sigmoidf_(acc[0][j][r] + bi);
      const float fv = sigmoidf_(acc[1][j][r] + bf);
      const float gv = tanhf    (acc[2][j][r] + bg);
      const float ov = sigmoidf_(acc[3][j][r] + bo);
      const float c_old = c_state[idx];
      const float h_old = h_state[idx];
      const float c_new = fv * c_old + iv * gv;
      const float h_new = ov * tanhf(c_new);
      const float c = m ? c_new : c_old;
      const float h = m ? h_new : h_old;
      c_state[idx] = c;
      h_state[idx] = h;
      hs_t[idx] = (bf16_t)h;                   // post-mask h: scan record AND next-step input
    }
  }
}

// ---------------- logits head: [T*B,1024] @ [1024,32]^T + b_out, masked ----------------
__global__ __launch_bounds__(256) void logits_kernel(
    const bf16_t* __restrict__ hs,     // [SEQ, BAT, HID] final layer
    const bf16_t* __restrict__ Wout,   // [VOC, HID] row-major bf16
    const float*  __restrict__ bout,   // [VOC]
    const int*    __restrict__ lengths,
    float* __restrict__ out)           // [BAT, SEQ, VOC]
{
  const int lane  = threadIdx.x & 31;
  const int wave  = threadIdx.x >> 5;
  const int wid   = blockIdx.x * 8 + wave;    // 0..2047
  const int mbase = wid * 16;                 // row tile in T*B

  f32x8 acc[2];
  acc[0] = f32x8{0.f,0.f,0.f,0.f,0.f,0.f,0.f,0.f};
  acc[1] = acc[0];

  for (int kt = 0; kt < 32; ++kt) {
    const int k = kt * 32;
    if (kt + 1 < 32)
      __builtin_prefetch(hs + (size_t)(mbase + (lane & 15)) * HID + k + 32, 0, 3);
    bf16x16 a = load_a_frag(hs, HID, mbase, k, lane);
    #pragma unroll
    for (int j = 0; j < 2; ++j) {
      bf16x16 bb = load_b_frag(Wout, HID, j * 16, k, lane);
      acc[j] = wmma_bf16(a, bb, acc[j]);
    }
  }

  const int half = lane >> 4;
  const int ncol = lane & 15;
  #pragma unroll
  for (int j = 0; j < 2; ++j) {
    const int v = j * 16 + ncol;
    const float bv = bout[v];
    #pragma unroll
    for (int r = 0; r < 8; ++r) {
      const int row = mbase + half * 8 + r;   // row = t*BAT + b
      const int t = row >> 8;                 // / BAT
      const int b = row & (BAT - 1);
      float val = acc[j][r] + bv;
      if (t >= lengths[b]) val = 0.f;
      out[(size_t)b * SEQ * VOC + (size_t)t * VOC + v] = val;
    }
  }
}

// copy final fp32 h/c states into d_out after the logits block
__global__ void copy_states_kernel(const float* __restrict__ h_state,
                                   const float* __restrict__ c_state,
                                   float* __restrict__ out) {
  const int n = NLAY * BAT * HID;                    // 786432
  const size_t off = (size_t)BAT * SEQ * VOC;        // logits elements
  for (int i = blockIdx.x * blockDim.x + threadIdx.x; i < n; i += gridDim.x * blockDim.x) {
    out[off + i]     = h_state[i];
    out[off + n + i] = c_state[i];
  }
}

// ---------------- host side ----------------
extern "C" void kernel_launch(void* const* d_in, const int* in_sizes, int n_in,
                              void* d_out, int out_size, void* d_ws, size_t ws_size,
                              hipStream_t stream) {
  const int*   tokens    = (const int*)  d_in[0];
  const int*   lengths   = (const int*)  d_in[1];
  const float* W_ih0     = (const float*)d_in[2];   // [4H, V]
  const float* W_hh0     = (const float*)d_in[3];   // [4H, H]
  const float* b0        = (const float*)d_in[4];   // [4H]
  const float* W_ih_rest = (const float*)d_in[5];   // [2, 4H, H]
  const float* W_hh_rest = (const float*)d_in[6];   // [2, 4H, H]
  const float* b_rest    = (const float*)d_in[7];   // [2, 4H]
  const float* W_out     = (const float*)d_in[8];   // [V, H]
  const float* b_out     = (const float*)d_in[9];   // [V]
  float* out = (float*)d_out;

  // ---- workspace layout ----
  char* ws = (char*)d_ws;
  size_t off = 0;
  auto take = [&](size_t bytes) { char* p = ws + off; off += (bytes + 255) & ~(size_t)255; return p; };

  bf16_t* Wih0_bf = (bf16_t*)take((size_t)G4H * VOC * 2);          // 256 KB
  bf16_t* Wih_bf  = (bf16_t*)take((size_t)2 * G4H * HID * 2);      // 16 MB (layers 1,2)
  bf16_t* Whh_bf  = (bf16_t*)take((size_t)3 * G4H * HID * 2);      // 24 MB
  bf16_t* Wout_bf = (bf16_t*)take((size_t)VOC * HID * 2);          // 64 KB
  bf16_t* x0      = (bf16_t*)take((size_t)SEQ * BAT * VOC * 2);    // 2 MB
  bf16_t* hsA     = (bf16_t*)take((size_t)SEQ * BAT * HID * 2);    // 64 MB
  bf16_t* hsB     = (bf16_t*)take((size_t)SEQ * BAT * HID * 2);    // 64 MB
  float*  h_state = (float*) take((size_t)NLAY * BAT * HID * 4);   // 3 MB
  float*  c_state = (float*) take((size_t)NLAY * BAT * HID * 4);   // 3 MB (contiguous with h)

  // ---- weight conversion fp32 -> bf16 ----
  f32_to_bf16_kernel<<<256,  256, 0, stream>>>(W_ih0,     Wih0_bf, G4H * VOC);
  f32_to_bf16_kernel<<<2048, 256, 0, stream>>>(W_ih_rest, Wih_bf,  2 * G4H * HID);
  f32_to_bf16_kernel<<<2048, 256, 0, stream>>>(W_hh0,     Whh_bf,  G4H * HID);
  f32_to_bf16_kernel<<<2048, 256, 0, stream>>>(W_hh_rest, Whh_bf + (size_t)G4H * HID, 2 * G4H * HID);
  f32_to_bf16_kernel<<<256,  256, 0, stream>>>(W_out,     Wout_bf, VOC * HID);

  // ---- masked one-hot input & zeroed fp32 states ----
  onehot_kernel<<<1024, 256, 0, stream>>>(tokens, lengths, x0);
  zero_f32_kernel<<<1024, 256, 0, stream>>>(h_state, 2 * NLAY * BAT * HID); // h & c contiguous

  // ---- sequential LSTM: layer-major, step chain within each layer ----
  for (int l = 0; l < NLAY; ++l) {
    const bf16_t* x_base = (l == 0) ? x0 : (l == 1 ? hsA : hsB);
    const int lda  = (l == 0) ? VOC : HID;
    const int nkt  = (l == 0) ? 1 : 32;
    bf16_t* hs_out = (l == 1) ? hsB : hsA;        // l0->A, l1->B, l2->A
    const bf16_t* Wih  = (l == 0) ? Wih0_bf : Wih_bf + (size_t)(l - 1) * G4H * HID;
    const bf16_t* Whh  = Whh_bf + (size_t)l * G4H * HID;
    const float*  bias = (l == 0) ? b0 : b_rest + (size_t)(l - 1) * G4H;
    float* hst = h_state + (size_t)l * BAT * HID;
    float* cst = c_state + (size_t)l * BAT * HID;
    for (int t = 0; t < SEQ; ++t) {
      lstm_step_kernel<<<64, 256, 0, stream>>>(
          x_base + (size_t)t * BAT * lda, lda, nkt,
          Wih, Whh, bias, hs_out, hst, cst, lengths, t);
    }
  }

  // ---- head + final states ----
  logits_kernel<<<256, 256, 0, stream>>>(hsA, Wout_bf, b_out, lengths, out);
  copy_states_kernel<<<1024, 256, 0, stream>>>(h_state, c_state, out);
}
